// SA_Layer_77824807404163
// MI455X (gfx1250) — matrix operand order repaired and checked by reading, not compiled, not run on hardware
//
#include <hip/hip_runtime.h>
#include <math.h>

#define BB   16
#define CC   256
#define NP   2048
#define DQKD 64
#define NBINS 32
#define RP   64      // rel-table rows padded 63 -> 64
#define BN_EPS 1e-5f

typedef __attribute__((ext_vector_type(16))) __bf16 v16bf;
typedef __attribute__((ext_vector_type(8)))  float  v8f;

union BF8pack { __bf16 h[8]; uint4 q; };

// ---- WMMA operand loaders (contiguous-K bf16 layout, same for A and B, wave32) -----------------
// Pointers are pre-offset per lane; only a constant kk is added in the loop.
// A 16x32 (MxK): lane m(0-15) holds row m, K={0..15}; lane m+16 holds
// row m, K={16..31}.
__device__ __forceinline__ v16bf load_a_at(const __bf16* p) {
  union { v16bf v; uint4 q[2]; } u;
  u.q[0] = *reinterpret_cast<const uint4*>(p);
  u.q[1] = *reinterpret_cast<const uint4*>(p + 8);
  return u.v;
}
// B 32x16 (KxN): lane holds column n=lane&15, 16 consecutive K from (lane>>4)*16.
__device__ __forceinline__ v16bf load_b_at(const __bf16* p) {
  union { v16bf v; uint4 q[2]; } u;
  u.q[0] = *reinterpret_cast<const uint4*>(p);
  u.q[1] = *reinterpret_cast<const uint4*>(p + 8);
  return u.v;
}
__device__ __forceinline__ v8f wmma_bf16(v16bf a, v16bf b, v8f c) {
  return __builtin_amdgcn_wmma_f32_16x16x32_bf16(false, a, false, b, (short)0, c,
                                                 false, false);
}

// ---- elementwise converts ---------------------------------------------------
__global__ void cvt_bf16_kernel(const float* __restrict__ in,
                                __bf16* __restrict__ out, int n) {
  int i = blockIdx.x * blockDim.x + threadIdx.x;
  if (i < n) out[i] = (__bf16)in[i];
}

__global__ void ltpad_kernel(const float* __restrict__ xlt,
                             const float* __restrict__ ylt,
                             const float* __restrict__ zlt,
                             __bf16* __restrict__ out) {
  int i = blockIdx.x * blockDim.x + threadIdx.x;         // [3][64][64]
  if (i >= 3 * RP * DQKD) return;
  int t = i / (RP * DQKD);
  int rem = i % (RP * DQKD);
  int r = rem / DQKD, q = rem % DQKD;
  const float* src = (t == 0) ? xlt : ((t == 1) ? ylt : zlt);
  float v = (r < 2 * NBINS - 1) ? src[r * DQKD + q] : 0.f;
  out[i] = (__bf16)v;
}

// xT[b][n][c] = x[b][c][n]  (f32 -> bf16, LDS tiled transpose)
__global__ void transpose_cvt_kernel(const float* __restrict__ x,
                                     __bf16* __restrict__ xT) {
  __shared__ float tile[32][33];
  int b = blockIdx.z;
  int c0 = blockIdx.x * 32, n0 = blockIdx.y * 32;
  int tx = threadIdx.x, ty = threadIdx.y;            // block (32,8)
#pragma unroll
  for (int j = 0; j < 4; ++j)
    tile[ty + 8 * j][tx] =
        x[((size_t)b * CC + c0 + ty + 8 * j) * NP + n0 + tx];
  __syncthreads();
#pragma unroll
  for (int j = 0; j < 4; ++j)
    xT[((size_t)b * NP + n0 + ty + 8 * j) * CC + c0 + tx] =
        (__bf16)tile[tx][ty + 8 * j];
}

// ---- generic NT GEMM: D[M,N] = A[M,K] * B[N,K]^T (+row bias), batched -------
// Wave computes a 16x64 strip (4 fixed N-tiles, named accumulators).
template <bool OUT_BF16>
__global__ void __launch_bounds__(128)
gemm_nt_kernel(const __bf16* __restrict__ A, const __bf16* __restrict__ Bm,
               void* __restrict__ D, const float* __restrict__ bias,
               int M, int K, int lda, int ldb, int ldd,
               long long sA, long long sB, long long sD) {
  int b = blockIdx.z;
  int lane = threadIdx.x & 31, wave = threadIdx.x >> 5;
  int ml = lane & 15, hi = lane >> 4;
  int mbase = (blockIdx.x * 4 + wave) * 16;
  if (mbase >= M) return;
  int nb0 = blockIdx.y * 64;
  const __bf16* ap  = A + (size_t)b * sA + (size_t)(mbase + ml) * lda + hi * 16;
  const __bf16* bp0 = Bm + (size_t)b * sB + (size_t)(nb0 + ml) * ldb + hi * 16;
  const __bf16* bp1 = bp0 + (size_t)16 * ldb;
  const __bf16* bp2 = bp0 + (size_t)32 * ldb;
  const __bf16* bp3 = bp0 + (size_t)48 * ldb;
  v8f acc0 = {}, acc1 = {}, acc2 = {}, acc3 = {};
#pragma unroll 2
  for (int kk = 0; kk < K; kk += 32) {
    v16bf a = load_a_at(ap + kk);
    acc0 = wmma_bf16(a, load_b_at(bp0 + kk), acc0);
    acc1 = wmma_bf16(a, load_b_at(bp1 + kk), acc1);
    acc2 = wmma_bf16(a, load_b_at(bp2 + kk), acc2);
    acc3 = wmma_bf16(a, load_b_at(bp3 + kk), acc3);
  }
  float bia[8];
#pragma unroll
  for (int v = 0; v < 8; ++v)
    bia[v] = bias ? bias[mbase + hi * 8 + v] : 0.f;
  size_t dbase = (size_t)b * sD + (size_t)(mbase + hi * 8) * ldd + nb0 + ml;
  auto st = [&](v8f acc, int nt) {
#pragma unroll
    for (int v = 0; v < 8; ++v) {
      float val = acc[v] + bia[v];
      size_t idx = dbase + (size_t)v * ldd + nt * 16;
      if (OUT_BF16) ((__bf16*)D)[idx] = (__bf16)val;
      else          ((float*)D)[idx]  = val;
    }
  };
  st(acc0, 0); st(acc1, 1); st(acc2, 2); st(acc3, 3);
}

// ---- energy pass 1: per-row max / sum-of-exp --------------------------------
__global__ void __launch_bounds__(128)
rowstats_kernel(const __bf16* __restrict__ Qb, const __bf16* __restrict__ Kb,
                const float* __restrict__ qlt, const int* __restrict__ discrete,
                float* __restrict__ rowmax, float* __restrict__ rowsum) {
  int b = blockIdx.y;
  int lane = threadIdx.x & 31, wave = threadIdx.x >> 5;
  int ml = lane & 15, hi = lane >> 4;
  int nbase = (blockIdx.x * 4 + wave) * 16;
  const __bf16* Q = Qb + (size_t)b * NP * DQKD;
  const __bf16* aq = Q + (size_t)(nbase + ml) * DQKD + hi * 16;
  v16bf a0 = load_a_at(aq);
  v16bf a1 = load_a_at(aq + 32);
  const __bf16* kp = Kb + (size_t)b * NP * DQKD + (size_t)ml * DQKD + hi * 16;
  const int* dp = discrete + ((size_t)b * NP + ml) * 3;
  int rbase = nbase + hi * 8;
  const float* qx = qlt + (size_t)(b * 3 + 0) * NP * RP;
  const float* qy = qlt + (size_t)(b * 3 + 1) * NP * RP;
  const float* qz = qlt + (size_t)(b * 3 + 2) * NP * RP;
  int dxn[8], dyn[8], dzn[8], noff[8];
#pragma unroll
  for (int v = 0; v < 8; ++v) {
    const int* dn = discrete + ((size_t)b * NP + rbase + v) * 3;
    dxn[v] = dn[0]; dyn[v] = dn[1]; dzn[v] = dn[2];
    noff[v] = (rbase + v) * RP;
  }
  float lmax[8], lsum[8];
#pragma unroll
  for (int v = 0; v < 8; ++v) { lmax[v] = -1e30f; lsum[v] = 0.f; }
  for (int mb = 0; mb < NP; mb += 16) {
    v16bf b0 = load_b_at(kp);
    v16bf b1 = load_b_at(kp + 32);
    kp += 16 * DQKD;
    v8f acc = {};
    acc = wmma_bf16(a0, b0, acc);
    acc = wmma_bf16(a1, b1, acc);
    int dxm = dp[0] + NBINS - 1, dym = dp[1] + NBINS - 1, dzm = dp[2] + NBINS - 1;
    dp += 16 * 3;
#pragma unroll
    for (int v = 0; v < 8; ++v) {
      float e = acc[v] + qx[noff[v] + (dxm - dxn[v])]
                       + qy[noff[v] + (dym - dyn[v])]
                       + qz[noff[v] + (dzm - dzn[v])];
      float nm = fmaxf(lmax[v], e);
      lsum[v] = lsum[v] * __expf(lmax[v] - nm) + __expf(e - nm);
      lmax[v] = nm;
    }
  }
#pragma unroll
  for (int v = 0; v < 8; ++v) {          // combine the 16 column-lanes
    float gm = lmax[v];
#pragma unroll
    for (int s = 1; s < 16; s <<= 1) gm = fmaxf(gm, __shfl_xor(gm, s, 32));
    float ps = lsum[v] * __expf(lmax[v] - gm);
#pragma unroll
    for (int s = 1; s < 16; s <<= 1) ps += __shfl_xor(ps, s, 32);
    if (ml == 0) {
      rowmax[(size_t)b * NP + rbase + v] = gm;
      rowsum[(size_t)b * NP + rbase + v] = ps;
    }
  }
}

// ---- energy pass 2: write normalized attn^T (bf16) + column sums ------------
__global__ void __launch_bounds__(128)
attn_kernel(const __bf16* __restrict__ Qb, const __bf16* __restrict__ Kb,
            const float* __restrict__ qlt, const int* __restrict__ discrete,
            const float* __restrict__ rowmax, const float* __restrict__ rowsum,
            __bf16* __restrict__ attnT, float* __restrict__ colsum) {
  int b = blockIdx.y;
  int lane = threadIdx.x & 31, wave = threadIdx.x >> 5;
  int ml = lane & 15, hi = lane >> 4;
  int nbase = (blockIdx.x * 4 + wave) * 16;
  const __bf16* Q = Qb + (size_t)b * NP * DQKD;
  const __bf16* aq = Q + (size_t)(nbase + ml) * DQKD + hi * 16;
  v16bf a0 = load_a_at(aq);
  v16bf a1 = load_a_at(aq + 32);
  const __bf16* kp = Kb + (size_t)b * NP * DQKD + (size_t)ml * DQKD + hi * 16;
  const int* dp = discrete + ((size_t)b * NP + ml) * 3;
  int rbase = nbase + hi * 8;
  const float* qx = qlt + (size_t)(b * 3 + 0) * NP * RP;
  const float* qy = qlt + (size_t)(b * 3 + 1) * NP * RP;
  const float* qz = qlt + (size_t)(b * 3 + 2) * NP * RP;
  int dxn[8], dyn[8], dzn[8], noff[8];
  float rm[8], ri[8];
#pragma unroll
  for (int v = 0; v < 8; ++v) {
    const int* dn = discrete + ((size_t)b * NP + rbase + v) * 3;
    dxn[v] = dn[0]; dyn[v] = dn[1]; dzn[v] = dn[2];
    noff[v] = (rbase + v) * RP;
    rm[v] = rowmax[(size_t)b * NP + rbase + v];
    ri[v] = 1.f / rowsum[(size_t)b * NP + rbase + v];
  }
  __bf16* sp = attnT + (size_t)b * NP * NP + (size_t)ml * NP + rbase;
  float* cp = colsum + (size_t)b * NP + ml;
  for (int mb = 0; mb < NP; mb += 16) {
    v16bf b0 = load_b_at(kp);
    v16bf b1 = load_b_at(kp + 32);
    kp += 16 * DQKD;
    v8f acc = {};
    acc = wmma_bf16(a0, b0, acc);
    acc = wmma_bf16(a1, b1, acc);
    int dxm = dp[0] + NBINS - 1, dym = dp[1] + NBINS - 1, dzm = dp[2] + NBINS - 1;
    dp += 16 * 3;
    BF8pack pk;
    float part = 0.f;
#pragma unroll
    for (int v = 0; v < 8; ++v) {
      float e = acc[v] + qx[noff[v] + (dxm - dxn[v])]
                       + qy[noff[v] + (dym - dyn[v])]
                       + qz[noff[v] + (dzm - dzn[v])];
      float at = __expf(e - rm[v]) * ri[v];
      pk.h[v] = (__bf16)at;
      part += at;
    }
    *reinterpret_cast<uint4*>(sp) = pk.q;        // attnT[m][rbase..rbase+7]
    sp += (size_t)16 * NP;
    part += __shfl_xor(part, 16, 32);
    if (hi == 0) atomicAdd(cp, part);
    cp += 16;
  }
}

// ---- x_r = V @ attn / colsum ; diffT[m][d] = x[d][m] - x_r ------------------
__global__ void __launch_bounds__(128)
xr_diff_kernel(const __bf16* __restrict__ Vb, const __bf16* __restrict__ attnT,
               const float* __restrict__ colsum, const float* __restrict__ x,
               __bf16* __restrict__ diffT) {
  int b = blockIdx.z;
  int lane = threadIdx.x & 31, wave = threadIdx.x >> 5;
  int ml = lane & 15, hi = lane >> 4;
  int dbase = (blockIdx.x * 4 + wave) * 16;
  int mb0 = blockIdx.y * 64;
  const __bf16* ap  = Vb + (size_t)b * CC * NP + (size_t)(dbase + ml) * NP + hi * 16;
  const __bf16* bp0 = attnT + (size_t)b * NP * NP + (size_t)(mb0 + ml) * NP + hi * 16;
  const __bf16* bp1 = bp0 + (size_t)16 * NP;
  const __bf16* bp2 = bp0 + (size_t)32 * NP;
  const __bf16* bp3 = bp0 + (size_t)48 * NP;
  v8f acc0 = {}, acc1 = {}, acc2 = {}, acc3 = {};
#pragma unroll 2
  for (int kk = 0; kk < NP; kk += 32) {
    v16bf a = load_a_at(ap + kk);
    acc0 = wmma_bf16(a, load_b_at(bp0 + kk), acc0);
    acc1 = wmma_bf16(a, load_b_at(bp1 + kk), acc1);
    acc2 = wmma_bf16(a, load_b_at(bp2 + kk), acc2);
    acc3 = wmma_bf16(a, load_b_at(bp3 + kk), acc3);
  }
  const float* xp = x + ((size_t)b * CC + dbase + hi * 8) * NP;
  const float* cs = colsum + (size_t)b * NP;
  __bf16* dT = diffT + (size_t)b * NP * CC + dbase + hi * 8;
  auto st = [&](v8f acc, int nt) {
    int m = mb0 + nt * 16 + ml;
    float inv = 1.f / (1e-9f + cs[m]);
    BF8pack pk;
#pragma unroll
    for (int v = 0; v < 8; ++v)
      pk.h[v] = (__bf16)(xp[(size_t)v * NP + m] - acc[v] * inv);
    *reinterpret_cast<uint4*>(dT + (size_t)m * CC) = pk.q;
  };
  st(acc0, 0); st(acc1, 1); st(acc2, 2); st(acc3, 3);
}

// ---- y = Wt@diff + bt ; BN(eval) ; ReLU ; out = x + y -----------------------
__global__ void __launch_bounds__(128)
final_kernel(const __bf16* __restrict__ Wtb, const __bf16* __restrict__ diffT,
             const float* __restrict__ bt, const float* __restrict__ gamma,
             const float* __restrict__ beta, const float* __restrict__ mean,
             const float* __restrict__ var, const float* __restrict__ x,
             float* __restrict__ out) {
  int b = blockIdx.z;
  int lane = threadIdx.x & 31, wave = threadIdx.x >> 5;
  int ml = lane & 15, hi = lane >> 4;
  int obase = (blockIdx.x * 4 + wave) * 16;
  int mb0 = blockIdx.y * 64;
  const __bf16* ap  = Wtb + (size_t)(obase + ml) * CC + hi * 16;
  const __bf16* bp0 = diffT + (size_t)b * NP * CC + (size_t)(mb0 + ml) * CC + hi * 16;
  const __bf16* bp1 = bp0 + (size_t)16 * CC;
  const __bf16* bp2 = bp0 + (size_t)32 * CC;
  const __bf16* bp3 = bp0 + (size_t)48 * CC;
  v8f acc0 = {}, acc1 = {}, acc2 = {}, acc3 = {};
#pragma unroll 2
  for (int kk = 0; kk < CC; kk += 32) {
    v16bf a = load_a_at(ap + kk);
    acc0 = wmma_bf16(a, load_b_at(bp0 + kk), acc0);
    acc1 = wmma_bf16(a, load_b_at(bp1 + kk), acc1);
    acc2 = wmma_bf16(a, load_b_at(bp2 + kk), acc2);
    acc3 = wmma_bf16(a, load_b_at(bp3 + kk), acc3);
  }
  float btv[8], invv[8], mnv[8], bev[8];
#pragma unroll
  for (int v = 0; v < 8; ++v) {
    int o = obase + hi * 8 + v;
    btv[v] = bt[o];
    invv[v] = gamma[o] * rsqrtf(var[o] + BN_EPS);
    mnv[v] = mean[o];
    bev[v] = beta[o];
  }
  const float* xp = x + ((size_t)b * CC + obase + hi * 8) * NP;
  float* op = out + ((size_t)b * CC + obase + hi * 8) * NP;
  auto st = [&](v8f acc, int nt) {
    int m = mb0 + nt * 16 + ml;
#pragma unroll
    for (int v = 0; v < 8; ++v) {
      float val = acc[v] + btv[v];
      val = (val - mnv[v]) * invv[v] + bev[v];
      val = fmaxf(val, 0.f);
      op[(size_t)v * NP + m] = xp[(size_t)v * NP + m] + val;
    }
  };
  st(acc0, 0); st(acc1, 1); st(acc2, 2); st(acc3, 3);
}

// ---------------------------------------------------------------------------
extern "C" void kernel_launch(void* const* d_in, const int* in_sizes, int n_in,
                              void* d_out, int out_size, void* d_ws,
                              size_t ws_size, hipStream_t stream) {
  (void)in_sizes; (void)n_in; (void)out_size; (void)ws_size;
  const float* x        = (const float*)d_in[0];
  const int*   discrete = (const int*)d_in[1];
  /* d_in[2] = xyz: unused by reference */
  const float* Wq   = (const float*)d_in[3];
  const float* Wk   = (const float*)d_in[4];
  const float* Wv   = (const float*)d_in[5];
  const float* bv   = (const float*)d_in[6];
  const float* xlt  = (const float*)d_in[7];
  const float* ylt  = (const float*)d_in[8];
  const float* zlt  = (const float*)d_in[9];
  const float* Wt   = (const float*)d_in[10];
  const float* bt   = (const float*)d_in[11];
  const float* gam  = (const float*)d_in[12];
  const float* bet  = (const float*)d_in[13];
  const float* rmn  = (const float*)d_in[14];
  const float* rvr  = (const float*)d_in[15];
  float* out = (float*)d_out;

  char* ws = (char*)d_ws;
  size_t off = 0;
  auto take = [&](size_t bytes) -> char* {
    off = (off + 255) & ~(size_t)255;
    char* p = ws + off;
    off += bytes;
    return p;
  };
  __bf16* Wq_b  = (__bf16*)take((size_t)DQKD * CC * 2);
  __bf16* Wk_b  = (__bf16*)take((size_t)DQKD * CC * 2);
  __bf16* Wv_b  = (__bf16*)take((size_t)CC * CC * 2);
  __bf16* Wt_b  = (__bf16*)take((size_t)CC * CC * 2);
  __bf16* lt_b  = (__bf16*)take((size_t)3 * RP * DQKD * 2);
  __bf16* xT    = (__bf16*)take((size_t)BB * NP * CC * 2);
  __bf16* Qb    = (__bf16*)take((size_t)BB * NP * DQKD * 2);
  __bf16* Kb    = (__bf16*)take((size_t)BB * NP * DQKD * 2);
  __bf16* Vb    = (__bf16*)take((size_t)BB * CC * NP * 2);
  float*  qlt   = (float*)take((size_t)BB * 3 * NP * RP * 4);
  float*  rowmax= (float*)take((size_t)BB * NP * 4);
  float*  rowsum= (float*)take((size_t)BB * NP * 4);
  float*  colsum= (float*)take((size_t)BB * NP * 4);
  __bf16* attnT = (__bf16*)take((size_t)BB * NP * NP * 2);
  __bf16* diffT = (__bf16*)take((size_t)BB * NP * CC * 2);

  // weight / table converts
  cvt_bf16_kernel<<<(DQKD * CC + 255) / 256, 256, 0, stream>>>(Wq, Wq_b, DQKD * CC);
  cvt_bf16_kernel<<<(DQKD * CC + 255) / 256, 256, 0, stream>>>(Wk, Wk_b, DQKD * CC);
  cvt_bf16_kernel<<<(CC * CC + 255) / 256, 256, 0, stream>>>(Wv, Wv_b, CC * CC);
  cvt_bf16_kernel<<<(CC * CC + 255) / 256, 256, 0, stream>>>(Wt, Wt_b, CC * CC);
  ltpad_kernel<<<(3 * RP * DQKD + 255) / 256, 256, 0, stream>>>(xlt, ylt, zlt, lt_b);

  // xT[b][n][c]
  transpose_cvt_kernel<<<dim3(CC / 32, NP / 32, BB), dim3(32, 8), 0, stream>>>(x, xT);

  // Q = xT @ Wq^T  [N,64]; K likewise
  gemm_nt_kernel<true><<<dim3(NP / 64, 1, BB), 128, 0, stream>>>(
      xT, Wq_b, Qb, nullptr, NP, CC, CC, CC, DQKD,
      (long long)NP * CC, 0, (long long)NP * DQKD);
  gemm_nt_kernel<true><<<dim3(NP / 64, 1, BB), 128, 0, stream>>>(
      xT, Wk_b, Kb, nullptr, NP, CC, CC, CC, DQKD,
      (long long)NP * CC, 0, (long long)NP * DQKD);
  // V = Wv @ x + bv  stored [C,N]
  gemm_nt_kernel<true><<<dim3(CC / 64, NP / 64, BB), 128, 0, stream>>>(
      Wv_b, xT, Vb, bv, CC, CC, CC, CC, NP,
      0, (long long)NP * CC, (long long)CC * NP);
  // qlt[t] = Q @ lt^T  [N,64] f32, layout [b][t][n][r]
  for (int t = 0; t < 3; ++t) {
    gemm_nt_kernel<false><<<dim3(NP / 64, 1, BB), 128, 0, stream>>>(
        Qb, lt_b + (size_t)t * RP * DQKD, qlt + (size_t)t * NP * RP, nullptr,
        NP, DQKD, DQKD, DQKD, RP,
        (long long)NP * DQKD, 0, (long long)3 * NP * RP);
  }

  // softmax row statistics (pass 1)
  rowstats_kernel<<<dim3(NP / 64, BB), 128, 0, stream>>>(
      Qb, Kb, qlt, discrete, rowmax, rowsum);

  // attn (pass 2): write attnT bf16 + column sums
  hipMemsetAsync(colsum, 0, (size_t)BB * NP * 4, stream);
  attn_kernel<<<dim3(NP / 64, BB), 128, 0, stream>>>(
      Qb, Kb, qlt, discrete, rowmax, rowsum, attnT, colsum);

  // x_r GEMM + diff
  xr_diff_kernel<<<dim3(CC / 64, NP / 64, BB), 128, 0, stream>>>(
      Vb, attnT, colsum, x, diffT);

  // Wt GEMM + BN + ReLU + residual
  final_kernel<<<dim3(CC / 64, NP / 64, BB), 128, 0, stream>>>(
      Wt_b, diffT, bt, gam, bet, rmn, rvr, x, out);
}
